// ICouplingBlock_3487513444880
// MI455X (gfx1250) — compile-verified, matrix operand order checked by reading, not run on hardware
//
#include <hip/hip_runtime.h>
#include <math.h>

// ---------------------------------------------------------------------------
// ICouplingBlock for MI455X (gfx1250): layer-1 [68->128] GEMMs on WMMA f16
// matrix pipes (K=64 condition part), rank-4 sinusoidal part + activation on
// VALU, tiny [192->2] layer-2 on VALU. 128 rows per workgroup, 8 waves,
// one 16-row WMMA M-tile per wave, A-fragments of the shared condition
// block reused across both coupling branches.
// ---------------------------------------------------------------------------

typedef __attribute__((ext_vector_type(16))) _Float16 v16h;
typedef __attribute__((ext_vector_type(8)))  _Float16 v8h;
typedef __attribute__((ext_vector_type(8)))  float    v8f;

#define ROWS_PER_WG 128
#define CSTR  72    // padded f16 K-stride for [row][k<64] tiles (144B rows -> conflict-free b128)
#define HSTR  136   // padded f16 N-stride for hidden tile [row][n<128]
#define W2STR 200   // padded f32 row stride for layer-2 weights (dodges 2-way bank conflict)

#define SCALE1 0.12126781251816648f  // 1/sqrt(68)
#define SCALE2 0.07216878364870323f  // 1/sqrt(192)
#define SQRT2F 1.4142135623730951f

// LDS budget (bytes): 2*(3*128*CSTR + 128*HSTR) + 4*floats
static constexpr int SMEM_HALFS  = 3 * 128 * CSTR + 128 * HSTR;                 // 45056
static constexpr int SMEM_FLOATS = 512 + 512 + 2 * W2STR * 2 + 256 + 8 + 512 + 512 + 256; // 3368
static constexpr int SMEM_BYTES  = SMEM_HALFS * 2 + SMEM_FLOATS * 4;            // 103584

__device__ __forceinline__ v8f wmma16x16x32(v16h a, v16h b, v8f c) {
  // (neg_a, A, neg_b, B, c_mod, C, reuse_a, reuse_b)
  return __builtin_amdgcn_wmma_f32_16x16x32_f16(false, a, false, b, (short)0, c, false, false);
}

// Load one 16-half fragment per ISA layout: elems 0..7 = K kb..kb+7,
// elems 8..15 = K kb+16..kb+23 (caller folds kb into p). Two ds_load_b128.
__device__ __forceinline__ v16h load_frag(const _Float16* p) {
  v8h lo = *(const v8h*)(p);
  v8h hi = *(const v8h*)(p + 16);
  return __builtin_shufflevector(lo, hi, 0, 1, 2, 3, 4, 5, 6, 7,
                                         8, 9, 10, 11, 12, 13, 14, 15);
}

__device__ __forceinline__ void coupling_branch(
    const _Float16* __restrict__ cf,    // [128][CSTR] f16 condition (scaled to f16 verbatim)
    const _Float16* __restrict__ wb,    // [128 n][CSTR k] f16 W1 cond part * SCALE1
    const float*    __restrict__ ws,    // [128 n][4]  f32 W1 sin part * SCALE1
    const float*    __restrict__ b1,    // [128] layer-1 bias
    const float*    __restrict__ w2,    // [2][W2STR] f32 W2 * SCALE2
    const float*    __restrict__ b2,    // [2] layer-2 bias
    const float*    __restrict__ drive, // [128] coordinate supplying sinusoidals
    float*          __restrict__ upd,   // [128] coordinate updated in place
    float*          __restrict__ logs,  // [128] log-scale out
    float*          __restrict__ sinb,  // [128][4] scratch
    _Float16*       __restrict__ hbuf,  // [128][HSTR] scratch
    float*          __restrict__ obuf,  // [128][2] scratch
    v16h a0, v16h a1, int tid)
{
  const int lane = tid & 31;
  const int wave = tid >> 5;
  const int ln   = lane & 15;
  const int hi8  = (lane >> 4) << 3;

  // sinusoidal features of the driving coordinate: [sin t, sin 2t, cos t, cos 2t], t = v/10
  if (tid < ROWS_PER_WG) {
    float t = drive[tid] * 0.1f;
    sinb[tid * 4 + 0] = __sinf(t);
    sinb[tid * 4 + 1] = __sinf(t + t);
    sinb[tid * 4 + 2] = __cosf(t);
    sinb[tid * 4 + 3] = __cosf(t + t);
  }
  __syncthreads();

  // ---- layer 1: WMMA over K=64 condition block (2 k-steps x 8 n-tiles) ----
  v8f acc[8];
  const v8f vzero = {0.f, 0.f, 0.f, 0.f, 0.f, 0.f, 0.f, 0.f};
#pragma unroll
  for (int nt = 0; nt < 8; ++nt) acc[nt] = vzero;

#pragma unroll
  for (int nt = 0; nt < 8; ++nt) {
    const _Float16* bbase = wb + (nt * 16 + ln) * CSTR + hi8; // per-lane column of B
    v16h bf0 = load_frag(bbase);        // K = 0..31
    acc[nt] = wmma16x16x32(a0, bf0, acc[nt]);
    v16h bf1 = load_frag(bbase + 32);   // K = 32..63
    acc[nt] = wmma16x16x32(a1, bf1, acc[nt]);
  }

  // hoist sin features for this lane-half's 8 rows (broadcast LDS reads)
  float sf[8][4];
#pragma unroll
  for (int r = 0; r < 8; ++r) {
    const int m = wave * 16 + hi8 + r;
#pragma unroll
    for (int k = 0; k < 4; ++k) sf[r][k] = sinb[m * 4 + k];
  }

  // rank-4 sin part + bias + fused leaky-relu * sqrt(2); h -> LDS f16
#pragma unroll
  for (int nt = 0; nt < 8; ++nt) {
    const int n = nt * 16 + ln;
    const float w0 = ws[n * 4 + 0], w1 = ws[n * 4 + 1];
    const float w2s = ws[n * 4 + 2], w3 = ws[n * 4 + 3];
    const float bias = b1[n];
#pragma unroll
    for (int r = 0; r < 8; ++r) {
      float v = acc[nt][r] + sf[r][0] * w0 + sf[r][1] * w1
                          + sf[r][2] * w2s + sf[r][3] * w3 + bias;
      v = (v > 0.f ? v : 0.2f * v) * SQRT2F;
      hbuf[(wave * 16 + hi8 + r) * HSTR + n] = (_Float16)v;
    }
  }
  __syncthreads();

  // ---- layer 2: o[j] = [h(128), c(64)] . W2[j] * SCALE2 + b2[j]; 2 threads/row ----
  {
    const int row = tid >> 1;
    const int j   = tid & 1;
    const float* wr = w2 + j * W2STR;
    float dot = 0.f;
    const v8h* hp = (const v8h*)(hbuf + row * HSTR);
#pragma unroll
    for (int k8 = 0; k8 < 16; ++k8) {
      v8h hv = hp[k8];
#pragma unroll
      for (int i = 0; i < 8; ++i) dot += (float)hv[i] * wr[k8 * 8 + i];
    }
    const v8h* cp = (const v8h*)(cf + row * CSTR);
#pragma unroll
    for (int k8 = 0; k8 < 8; ++k8) {
      v8h cv = cp[k8];
#pragma unroll
      for (int i = 0; i < 8; ++i) dot += (float)cv[i] * wr[128 + k8 * 8 + i];
    }
    obuf[tid] = dot + b2[j];
  }
  __syncthreads();

  // affine coupling update
  if (tid < ROWS_PER_WG) {
    float ls = fminf(5.f, fmaxf(-5.f, obuf[tid * 2 + 0]));
    upd[tid] = upd[tid] * __expf(ls) + obuf[tid * 2 + 1];
    logs[tid] = ls;
  }
  __syncthreads();
}

__global__ __launch_bounds__(256) void icoupling_kernel(
    const float* __restrict__ coords, const float* __restrict__ cond,
    const float* __restrict__ Wx1, const float* __restrict__ bx1,
    const float* __restrict__ Wx2, const float* __restrict__ bx2,
    const float* __restrict__ Wy1, const float* __restrict__ by1,
    const float* __restrict__ Wy2, const float* __restrict__ by2,
    float* __restrict__ out, int nrows)
{
  extern __shared__ __align__(16) char smem_raw[];
  _Float16* cf   = (_Float16*)smem_raw;        // [128][CSTR]
  _Float16* wbx  = cf  + 128 * CSTR;           // [128][CSTR]
  _Float16* wby  = wbx + 128 * CSTR;           // [128][CSTR]
  _Float16* hbuf = wby + 128 * CSTR;           // [128][HSTR]
  float* fp   = (float*)(hbuf + 128 * HSTR);
  float* wsx  = fp; fp += 512;
  float* wsy  = fp; fp += 512;
  float* w2x  = fp; fp += 2 * W2STR;
  float* w2y  = fp; fp += 2 * W2STR;
  float* b1x  = fp; fp += 128;
  float* b1y  = fp; fp += 128;
  float* b2x  = fp; fp += 4;
  float* b2y  = fp; fp += 4;
  float* sinb = fp; fp += 512;
  float* xc   = fp; fp += 128;
  float* yc   = fp; fp += 128;
  float* xls  = fp; fp += 128;
  float* yls  = fp; fp += 128;
  float* obuf = fp; fp += 256;

  const int tid = threadIdx.x;
  const int g0  = blockIdx.x * ROWS_PER_WG;
  if (g0 >= nrows) return;

  // ---- stage condition tile f32 -> f16 (padded rows), coalesced float4 ----
  {
    const float4* c4 = (const float4*)(cond + (size_t)g0 * 64);
    for (int i = tid; i < ROWS_PER_WG * 16; i += 256) {
      float4 v = c4[i];
      _Float16* d = cf + (i >> 4) * CSTR + (i & 15) * 4;
      d[0] = (_Float16)v.x; d[1] = (_Float16)v.y;
      d[2] = (_Float16)v.z; d[3] = (_Float16)v.w;
    }
  }
  // ---- stage layer-1 weights: split sin (f32) / cond (f16), fold SCALE1 ----
  for (int i = tid; i < 128 * 68; i += 256) {
    const int n = i / 68, k = i - n * 68;
    const float vx = Wx1[i] * SCALE1;
    const float vy = Wy1[i] * SCALE1;
    if (k < 4) { wsx[n * 4 + k] = vx; wsy[n * 4 + k] = vy; }
    else       { wbx[n * CSTR + (k - 4)] = (_Float16)vx;
                 wby[n * CSTR + (k - 4)] = (_Float16)vy; }
  }
  // ---- stage layer-2 weights (fold SCALE2) and biases ----
  for (int i = tid; i < 2 * 192; i += 256) {
    const int j = i / 192, k = i - j * 192;
    w2x[j * W2STR + k] = Wx2[i] * SCALE2;
    w2y[j * W2STR + k] = Wy2[i] * SCALE2;
  }
  if (tid < 128) { b1x[tid] = bx1[tid]; b1y[tid] = by1[tid]; }
  if (tid < 2)   { b2x[tid] = bx2[tid]; b2y[tid] = by2[tid]; }
  if (tid < ROWS_PER_WG) {
    const float2 xy = ((const float2*)coords)[g0 + tid];
    xc[tid] = xy.x; yc[tid] = xy.y;
  }
  __syncthreads();

  // ---- per-lane A fragments of the shared condition block (ISA 16-bit A
  //      layout: lanes 0-15 K {0..7,16..23}, lanes 16-31 K {8..15,24..31});
  //      reused by both coupling branches ----
  const int lane = tid & 31, wave = tid >> 5;
  const int ln = lane & 15, hi8 = (lane >> 4) << 3;
  const _Float16* abase = cf + (wave * 16 + ln) * CSTR + hi8;
  v16h a0 = load_frag(abase);        // K = 0..31
  v16h a1 = load_frag(abase + 32);   // K = 32..63

  // branch 1: x drives, y updated.  branch 2: new y drives, x updated.
  coupling_branch(cf, wbx, wsx, b1x, w2x, b2x, xc, yc, yls, sinb, hbuf, obuf, a0, a1, tid);
  coupling_branch(cf, wby, wsy, b1y, w2y, b2y, yc, xc, xls, sinb, hbuf, obuf, a0, a1, tid);

  if (tid < ROWS_PER_WG) {
    const int g = g0 + tid;
    out[2 * (size_t)g + 0] = xc[tid];
    out[2 * (size_t)g + 1] = yc[tid];
    out[2 * (size_t)nrows + g] = xls[tid] + yls[tid];
  }
}

extern "C" void kernel_launch(void* const* d_in, const int* in_sizes, int n_in,
                              void* d_out, int out_size, void* d_ws, size_t ws_size,
                              hipStream_t stream) {
  (void)n_in; (void)out_size; (void)d_ws; (void)ws_size;
  const float* coords = (const float*)d_in[0];
  const float* cond   = (const float*)d_in[1];
  const float* Wx1    = (const float*)d_in[2];
  const float* bx1    = (const float*)d_in[3];
  const float* Wx2    = (const float*)d_in[4];
  const float* bx2    = (const float*)d_in[5];
  const float* Wy1    = (const float*)d_in[6];
  const float* by1    = (const float*)d_in[7];
  const float* Wy2    = (const float*)d_in[8];
  const float* by2    = (const float*)d_in[9];

  const int nrows = in_sizes[0] / 2;               // 524288 for B=64, N=8192
  const int grid  = nrows / ROWS_PER_WG;           // rows divisible by 128

  hipFuncSetAttribute((const void*)icoupling_kernel,
                      hipFuncAttributeMaxDynamicSharedMemorySize, SMEM_BYTES);
  icoupling_kernel<<<grid, 256, SMEM_BYTES, stream>>>(
      coords, cond, Wx1, bx1, Wx2, bx2, Wy1, by1, Wy2, by2,
      (float*)d_out, nrows);
}
